// NVFP4LinearKernel_33569464385924
// MI455X (gfx1250) — 8-bit matrix operand order repaired and checked by reading, compiled, not run on hardware
//
#include <hip/hip_runtime.h>


typedef __attribute__((ext_vector_type(16))) int   v16i;
typedef __attribute__((ext_vector_type(8)))  float v8f;

#define TILE        128
#define LDS_STRIDE  144

#if defined(__has_builtin)
#  if __has_builtin(__builtin_amdgcn_cvt_pk_fp8_f32)
#    define HAVE_HW_FP8_CVT 1
#  endif
#  if __has_builtin(__builtin_amdgcn_s_wait_asynccnt)
#    define HAVE_WAIT_ASYNC_BUILTIN 1
#  endif
#endif

typedef __attribute__((address_space(3))) void svoid;

__device__ __forceinline__ void wait_async_lds() {
#if defined(__AMDGCN__)
#  ifdef HAVE_WAIT_ASYNC_BUILTIN
  __builtin_amdgcn_s_wait_asynccnt(0);
#  else
  asm volatile("s_wait_asynccnt 0x0" ::: "memory");
#  endif
#endif
}

__device__ __forceinline__ void async_copy_b128(const unsigned char* gsrc,
                                                unsigned char* ldst) {
#if defined(__AMDGCN__)
  unsigned lds_addr =
      (unsigned)(unsigned long long)(svoid*)ldst;
  unsigned long long gaddr = (unsigned long long)gsrc;
  asm volatile("global_load_async_to_lds_b128 %0, %1, off"
               :
               : "v"(lds_addr), "v"(gaddr)
               : "memory");
#else
  *(uint4*)ldst = *(const uint4*)gsrc;
#endif
}

__device__ __forceinline__ unsigned char f32_to_e4m3_sw(float xf) {
  xf = fminf(448.f, fmaxf(-448.f, xf));
  unsigned bits = __float_as_uint(xf);
  unsigned sign = (bits >> 24) & 0x80u;
  unsigned ax   = bits & 0x7fffffffu;
  if (ax <= 0x3a800000u) return (unsigned char)sign;
  int e = (int)(ax >> 23) - 127;
  unsigned mant = ax & 0x7fffffu;
  unsigned vout;
  if (e < -6) {
    int rshift = 20 + (-6 - e);
    if (rshift > 24) return (unsigned char)sign;
    unsigned m24 = mant | 0x800000u;
    unsigned q   = m24 >> rshift;
    unsigned rem = m24 & ((1u << rshift) - 1u);
    unsigned hlf = 1u << (rshift - 1);
    if (rem > hlf || (rem == hlf && (q & 1u))) q++;
    vout = q;
  } else {
    unsigned q   = mant >> 20;
    unsigned rem = mant & 0xfffffu;
    if (rem > 0x80000u || (rem == 0x80000u && (q & 1u))) q++;
    vout = ((unsigned)(e + 7) << 3) + q;
    if (vout > 0x7eu) vout = 0x7eu;
  }
  return (unsigned char)(sign | vout);
}

__device__ __forceinline__ unsigned pack4_e4m3(float a, float b, float c, float d) {
#ifdef HAVE_HW_FP8_CVT
  int p = __builtin_amdgcn_cvt_pk_fp8_f32(a, b, 0, false);
  p     = __builtin_amdgcn_cvt_pk_fp8_f32(c, d, p, true);
  return (unsigned)p;
#else
  return (unsigned)f32_to_e4m3_sw(a)        |
         ((unsigned)f32_to_e4m3_sw(b) << 8) |
         ((unsigned)f32_to_e4m3_sw(c) << 16)|
         ((unsigned)f32_to_e4m3_sw(d) << 24);
#endif
}

__global__ void __launch_bounds__(256)
quant_fp8_kernel(const float* __restrict__ in, unsigned* __restrict__ out,
                 long long n4, const float* __restrict__ scale) {
  long long i = (long long)blockIdx.x * 256 + threadIdx.x;
  if (i >= n4) return;
  float m = scale ? (1.0f / scale[0]) : 1.0f;
  float4 v = ((const float4*)in)[i];
  float a = fminf(448.f, fmaxf(-448.f, v.x * m));
  float b = fminf(448.f, fmaxf(-448.f, v.y * m));
  float c = fminf(448.f, fmaxf(-448.f, v.z * m));
  float d = fminf(448.f, fmaxf(-448.f, v.w * m));
  out[i] = pack4_e4m3(a, b, c, d);
}

__device__ __forceinline__ v16i load_a_frag(const unsigned char* lds, int row, int g) {
  v16i a;
  const unsigned char* base = lds + row * LDS_STRIDE;
#pragma unroll
  for (int h = 0; h < 2; ++h) {
#pragma unroll
    for (int v = 0; v < 4; ++v) {
      uint2 d = *(const uint2*)(base + 64 * h + 16 * v + 8 * g);
      a[h * 8 + 2 * v]     = (int)d.x;
      a[h * 8 + 2 * v + 1] = (int)d.y;
    }
  }
  return a;
}

__device__ __forceinline__ v16i load_b_frag(const unsigned char* lds, int col, int g) {
  v16i b;
  const unsigned char* base = lds + col * LDS_STRIDE;
#pragma unroll
  for (int grp = 0; grp < 8; ++grp) {
    uint2 d = *(const uint2*)(base + 16 * grp + 8 * g);
    b[2 * grp + 0] = (int)d.x;
    b[2 * grp + 1] = (int)d.y;
  }
  return b;
}

__device__ __forceinline__ void stage_tiles(const unsigned char* __restrict__ gA,
                                            const unsigned char* __restrict__ gB,
                                            unsigned char* ldsA, unsigned char* ldsB,
                                            int rbase, int seg16, int K, int k0) {
#pragma unroll
  for (int p = 0; p < 4; ++p) {
    int row = 32 * p + rbase;
    size_t goff = (size_t)row * K + (size_t)(k0 + seg16);
    int loff = row * LDS_STRIDE + seg16;
    async_copy_b128(gA + goff, ldsA + loff);
    async_copy_b128(gB + goff, ldsB + loff);
  }
}

__global__ void __launch_bounds__(256)
fp8_gemm_kernel(const unsigned char* __restrict__ Aq,
                const unsigned char* __restrict__ Bq,
                const float* __restrict__ wscale,
                const float* __restrict__ iscale,
                const float* __restrict__ bias,
                float* __restrict__ out,
                int M, int N, int K) {
  __shared__ __align__(16) unsigned char lds_a[2][TILE * LDS_STRIDE];
  __shared__ __align__(16) unsigned char lds_b[2][TILE * LDS_STRIDE];

  const int t      = threadIdx.x;
  const int lane   = t & 31;
  const int w      = t >> 5;
  const int g      = lane >> 4;
  const int ln     = lane & 15;
  const int waveM  = (w >> 1) * 32;
  const int waveN  = (w & 1) * 64;
  const int blockN = blockIdx.x * TILE;
  const int blockM = blockIdx.y * TILE;

  const unsigned char* gA = Aq + (size_t)blockM * K;
  const unsigned char* gB = Bq + (size_t)blockN * K;
  const int rbase = t >> 3;
  const int seg16 = (t & 7) * 16;

  v8f acc[2][4];
#pragma unroll
  for (int i = 0; i < 2; ++i)
#pragma unroll
    for (int j = 0; j < 4; ++j)
      acc[i][j] = (v8f){0.f, 0.f, 0.f, 0.f, 0.f, 0.f, 0.f, 0.f};

  stage_tiles(gA, gB, lds_a[0], lds_b[0], rbase, seg16, K, 0);
  wait_async_lds();
  __syncthreads();

  const int nsteps = K / TILE;
  for (int kt = 0; kt < nsteps; ++kt) {
    const int cur = kt & 1;

    if (kt + 1 < nsteps)
      stage_tiles(gA, gB, lds_a[cur ^ 1], lds_b[cur ^ 1],
                  rbase, seg16, K, (kt + 1) * TILE);

    const unsigned char* la = lds_a[cur];
    const unsigned char* lb = lds_b[cur];

    v16i afrag[2];
#pragma unroll
    for (int i = 0; i < 2; ++i)
      afrag[i] = load_a_frag(la, waveM + 16 * i + ln, g);

    v16i bfrag[4];
#pragma unroll
    for (int j = 0; j < 4; ++j)
      bfrag[j] = load_b_frag(lb, waveN + 16 * j + ln, g);

#pragma unroll
    for (int i = 0; i < 2; ++i)
#pragma unroll
      for (int j = 0; j < 4; ++j)
        acc[i][j] = __builtin_amdgcn_wmma_f32_16x16x128_fp8_fp8(
            afrag[i], bfrag[j], (short)0, acc[i][j], false, false);

    wait_async_lds();
    __syncthreads();
  }

  const float s = iscale[0] * wscale[0];
#pragma unroll
  for (int i = 0; i < 2; ++i) {
#pragma unroll
    for (int j = 0; j < 4; ++j) {
      int col  = blockN + waveN + 16 * j + ln;
      int row0 = blockM + waveM + 16 * i + 8 * g;
      float bv = bias[col];
#pragma unroll
      for (int r = 0; r < 8; ++r)
        out[(size_t)(row0 + r) * N + col] = acc[i][j][r] * s + bv;
    }
  }
}

extern "C" void kernel_launch(void* const* d_in, const int* in_sizes, int n_in,
                              void* d_out, int out_size, void* d_ws, size_t ws_size,
                              hipStream_t stream) {
  const float* x      = (const float*)d_in[0];
  const float* wq     = (const float*)d_in[1];
  const float* wscale = (const float*)d_in[2];
  const float* iscale = (const float*)d_in[3];
  const float* bias   = (const float*)d_in[4];
  float* out = (float*)d_out;

  const int Dout = in_sizes[4];
  const int Din  = in_sizes[1] / Dout;
  const int T    = in_sizes[0] / Din;

  unsigned char* xq8 = (unsigned char*)d_ws;
  size_t woff = (((size_t)T * (size_t)Din) + 255) & ~(size_t)255;
  unsigned char* wq8 = (unsigned char*)d_ws + woff;

  long long n4x = (long long)T * Din / 4;
  long long n4w = (long long)Dout * Din / 4;

  quant_fp8_kernel<<<dim3((unsigned)((n4x + 255) / 256)), dim3(256), 0, stream>>>(
      x, (unsigned*)xq8, n4x, iscale);
  quant_fp8_kernel<<<dim3((unsigned)((n4w + 255) / 256)), dim3(256), 0, stream>>>(
      wq, (unsigned*)wq8, n4w, nullptr);

  dim3 grid((unsigned)(Dout / TILE), (unsigned)(T / TILE));
  fp8_gemm_kernel<<<grid, dim3(256), 0, stream>>>(
      xq8, wq8, wscale, iscale, bias, out, T, Dout, Din);
}
